// GeometricallyAlignedVmapAttention_16947940950547
// MI455X (gfx1250) — compile-verified, matrix operand order checked by reading, not run on hardware
//
#include <hip/hip_runtime.h>
#include <hip/hip_bf16.h>
#include <math.h>

#define EPSF 1e-7f
#define KNB 32

typedef __attribute__((ext_vector_type(16))) __bf16 v16bf;
typedef __attribute__((ext_vector_type(8)))  __bf16 v8bf;
typedef __attribute__((ext_vector_type(8)))  float  v8f;

static __device__ __forceinline__ unsigned short f32_to_bf16(float f) {
    unsigned int u = __float_as_uint(f);
    u += 0x7fffu + ((u >> 16) & 1u);   // round-to-nearest-even
    return (unsigned short)(u >> 16);
}

static __device__ __forceinline__ float waveAllSum(float v) {
#pragma unroll
    for (int m = 16; m > 0; m >>= 1) v += __shfl_xor(v, m, 32);
    return v;
}
static __device__ __forceinline__ float waveAllMax(float v) {
#pragma unroll
    for (int m = 16; m > 0; m >>= 1) v = fmaxf(v, __shfl_xor(v, m, 32));
    return v;
}

struct F3 { float x, y, z; };
static __device__ __forceinline__ float dot3(F3 a, F3 b) { return a.x*b.x + a.y*b.y + a.z*b.z; }

static __device__ __forceinline__ F3 mobius_add(F3 x, F3 y, float c, float sc) {
    float x2 = dot3(x, x), y2 = dot3(y, y), xy = dot3(x, y);
    float a   = 1.f + 2.f*c*xy + c*y2;
    float bb  = 1.f - c*x2;
    float den = fmaxf(1.f + 2.f*c*xy + c*c*x2*y2, EPSF);
    F3 r { (a*x.x + bb*y.x)/den, (a*x.y + bb*y.y)/den, (a*x.z + bb*y.z)/den };
    float nrm  = fmaxf(sqrtf(dot3(r, r)), EPSF);
    float maxn = (1.f - EPSF)/sc;
    float s = fminf(maxn/nrm, 1.f);
    r.x *= s; r.y *= s; r.z *= s;
    return r;
}

static __device__ __forceinline__ float hyp_dist(F3 x, F3 y, float c, float sc) {
    F3 ny { -y.x, -y.y, -y.z };
    F3 d = mobius_add(x, ny, c, sc);
    float dn = sqrtf(dot3(d, d));
    dn = fminf(fmaxf(dn, EPSF), 1.f - EPSF);
    float arg = fminf(sc*dn, 1.f - EPSF);
    return 2.f*atanhf(arg)/sc;
}

static __device__ __forceinline__ float gelu_tanh(float v) {
    float v3 = v*v*v;
    return 0.5f*v*(1.f + tanhf(0.7978845608028654f*(v + 0.044715f*v3)));
}

// ---------------- LayerNorm (f32 in -> bf16 out), one row per block ----------------
__global__ void ln_bf16_kernel(const float* __restrict__ in, const float* __restrict__ g,
                               const float* __restrict__ b, unsigned short* __restrict__ outBf,
                               int D) {
    __shared__ float ss[256], ss2[256];
    const int row = blockIdx.x;
    const float* x = in + (size_t)row * D;
    float s = 0.f, s2 = 0.f;
    for (int d = threadIdx.x; d < D; d += 256) { float v = x[d]; s += v; s2 += v*v; }
    ss[threadIdx.x] = s; ss2[threadIdx.x] = s2;
    __syncthreads();
    for (int st = 128; st > 0; st >>= 1) {
        if ((int)threadIdx.x < st) { ss[threadIdx.x] += ss[threadIdx.x+st]; ss2[threadIdx.x] += ss2[threadIdx.x+st]; }
        __syncthreads();
    }
    float mean = ss[0] / (float)D;
    float var  = ss2[0] / (float)D - mean*mean;
    float rstd = rsqrtf(var + 1e-6f);
    for (int d = threadIdx.x; d < D; d += 256) {
        float v = (x[d] - mean)*rstd*g[d] + b[d];
        outBf[(size_t)row*D + d] = f32_to_bf16(v);
    }
}

// ---------------- f32 (RxC) -> bf16 transposed (CxR) ----------------
__global__ void conv_transpose_bf16(const float* __restrict__ in, unsigned short* __restrict__ out,
                                    int R, int C) {
    int i = blockIdx.x*256 + threadIdx.x;
    if (i >= R*C) return;
    int r = i / C, cc = i % C;
    out[(size_t)cc*R + r] = f32_to_bf16(in[i]);
}

// ---------------- WMMA bf16 GEMM with async global->LDS double-buffered staging ------------
// out = f(A(MxK) * Bt(NxK)^T + bias) [+ residual]
// block = 128 threads (4 waves), block tile 64x64, wave tile 16x64 (4 accumulators).
// Per K-step: stage 64x32 A-tile (4KB) and 64x32 B-tile (4KB) into LDS with
// global_load_async_to_lds_b128 (ASYNCcnt), compute WMMA fragments out of LDS.
__global__ void gemm_bf16_wmma(const unsigned short* __restrict__ A,
                               const unsigned short* __restrict__ Bt,
                               const float* __restrict__ bias,
                               const float* __restrict__ residual,
                               float* __restrict__ outF,
                               unsigned short* __restrict__ outBf,
                               int M, int Nd, int Kd, int act) {
    __shared__ unsigned short sA[2][64*32];
    __shared__ unsigned short sB[2][64*32];

    const int tid  = threadIdx.x;
    const int lane = tid & 31;
    const int wave = tid >> 5;
    const int rowBlk = blockIdx.x*64;
    const int colBlk = blockIdx.y*64;
    const int lr = lane & 15;
    const int kh = (lane >> 4) << 3;   // lanes 0-15: K 0..7/16..23 ; lanes 16-31: K 8..15/24..31

    // stage one 64x32 A-tile + 64x32 B-tile (chunks of 16B; 128 threads x 2 chunks each side)
    auto stage = [&](int buf, int k0) {
#pragma unroll
        for (int s = 0; s < 2; ++s) {
            const int c = tid + s*128;          // 0..255
            const int r = c >> 2;               // tile row/col 0..63
            const int p = (c & 3) << 3;         // element offset within 32-K row: 0,8,16,24
            const unsigned short* ga = A  + (size_t)(rowBlk + r)*Kd + k0 + p;
            unsigned la = (unsigned)(unsigned long long)(&sA[buf][r*32 + p]);
            asm volatile("global_load_async_to_lds_b128 %0, %1, off"
                         :: "v"(la), "v"(ga) : "memory");
            const unsigned short* gb = Bt + (size_t)(colBlk + r)*Kd + k0 + p;
            unsigned lb = (unsigned)(unsigned long long)(&sB[buf][r*32 + p]);
            asm volatile("global_load_async_to_lds_b128 %0, %1, off"
                         :: "v"(lb), "v"(gb) : "memory");
        }
    };

    v8f acc[4] = {};

    stage(0, 0);
    asm volatile("s_wait_asynccnt 0x0" ::: "memory");
    __syncthreads();

    for (int k0 = 0; k0 < Kd; k0 += 32) {
        const int buf = (k0 >> 5) & 1;
        if (k0 + 32 < Kd) stage(buf ^ 1, k0 + 32);   // overlap copy of next tile with compute

        const unsigned short* sa = &sA[buf][(wave*16 + lr)*32 + kh];
        v8bf alo = *(const v8bf*)(sa);
        v8bf ahi = *(const v8bf*)(sa + 16);
        v16bf af = __builtin_shufflevector(alo, ahi, 0,1,2,3,4,5,6,7,8,9,10,11,12,13,14,15);
#pragma unroll
        for (int t = 0; t < 4; ++t) {
            const unsigned short* sb = &sB[buf][(t*16 + lr)*32 + kh];
            v8bf blo = *(const v8bf*)(sb);
            v8bf bhi = *(const v8bf*)(sb + 16);
            v16bf bfr = __builtin_shufflevector(blo, bhi, 0,1,2,3,4,5,6,7,8,9,10,11,12,13,14,15);
            acc[t] = __builtin_amdgcn_wmma_f32_16x16x32_bf16(
                false, af, false, bfr, (short)0, acc[t], false, false);
        }

        // next-tile copies complete + all waves done reading current buffer
        asm volatile("s_wait_asynccnt 0x0" ::: "memory");
        __syncthreads();
    }

    const int mb = rowBlk + wave*16 + ((lane >> 4) << 3);
#pragma unroll
    for (int t = 0; t < 4; ++t) {
        const int col = colBlk + t*16 + lr;
        const float bv = bias ? bias[col] : 0.f;
#pragma unroll
        for (int i = 0; i < 8; ++i) {
            const size_t off = (size_t)(mb + i)*Nd + col;
            float v = acc[t][i] + bv;
            if (act) v = gelu_tanh(v);
            if (residual) v += residual[off];
            if (outF)  outF[off]  = v;
            if (outBf) outBf[off] = f32_to_bf16(v);
        }
    }
}

// ---------------- top-K nearest (hyperbolic) neighbors per query ----------------
__global__ void topk_kernel(const float* __restrict__ pos, const float* __restrict__ cptr,
                            int* __restrict__ idxOut, int N) {
    __shared__ float sdist[2048];
    __shared__ float rv[256];
    __shared__ int   ri[256];
    const int bn = blockIdx.x;
    const int b = bn / N, n = bn % N;
    const float c = cptr[0];
    const float sc = fmaxf(sqrtf(c), EPSF);
    const float* pb = pos + (size_t)b*N*3;
    F3 qp { pb[n*3+0], pb[n*3+1], pb[n*3+2] };
    for (int j = threadIdx.x; j < N; j += 256) {
        F3 kp { pb[j*3+0], pb[j*3+1], pb[j*3+2] };
        sdist[j] = hyp_dist(qp, kp, c, sc);
    }
    __syncthreads();
    for (int kk = 0; kk < KNB; ++kk) {
        float bv = 3.4e38f; int bi = 0x7fffffff;
        for (int j = threadIdx.x; j < N; j += 256) {
            float dv = sdist[j];
            if (dv < bv || (dv == bv && j < bi)) { bv = dv; bi = j; }
        }
        rv[threadIdx.x] = bv; ri[threadIdx.x] = bi;
        __syncthreads();
        for (int st = 128; st > 0; st >>= 1) {
            if ((int)threadIdx.x < st) {
                float ov = rv[threadIdx.x+st]; int oi = ri[threadIdx.x+st];
                if (ov < rv[threadIdx.x] || (ov == rv[threadIdx.x] && oi < ri[threadIdx.x])) {
                    rv[threadIdx.x] = ov; ri[threadIdx.x] = oi;
                }
            }
            __syncthreads();
        }
        if (threadIdx.x == 0) {
            idxOut[(size_t)bn*KNB + kk] = ri[0];
            sdist[ri[0]] = 3.4e38f;
        }
        __syncthreads();
    }
}

// ---------------- geo = logmap0(mobius_add(-qpos, kpos, c), c), P=3 ----------------
__global__ void geo_kernel(const float* __restrict__ pos, const float* __restrict__ cptr,
                           const int* __restrict__ idx, float* __restrict__ geo,
                           int N, int total) {
    int t = blockIdx.x*256 + threadIdx.x;
    if (t >= total) return;
    int kk = t % KNB;
    int bn = t / KNB;
    int b = bn / N, n = bn % N;
    const float c = cptr[0];
    const float sc = fmaxf(sqrtf(c), EPSF);
    const float* pb = pos + (size_t)b*N*3;
    int j = idx[(size_t)bn*KNB + kk];
    F3 nq { -pb[n*3+0], -pb[n*3+1], -pb[n*3+2] };
    F3 kp {  pb[j*3+0],  pb[j*3+1],  pb[j*3+2] };
    F3 m = mobius_add(nq, kp, c, sc);
    float yn = sqrtf(dot3(m, m));
    float gx = 0.f, gy = 0.f, gz = 0.f;
    if (yn >= EPSF) {
        float syn = fmaxf(yn, EPSF);
        float arg = fminf(sc*syn, 1.f - EPSF);
        float s = (atanhf(arg)/sc) / syn;
        gx = m.x*s; gy = m.y*s; gz = m.z*s;
    }
    geo[(size_t)t*3+0] = gx;
    geo[(size_t)t*3+1] = gy;
    geo[(size_t)t*3+2] = gz;
}

// ---------------- gather-attention: one block per query, one wave32 per head ----------------
// align = q·(geo@Wg+bg) folded to (q@Wg^T)·geo + q·bg  — avoids materializing geo_v
__global__ void attn_kernel(const float* __restrict__ q, const float* __restrict__ k,
                            const float* __restrict__ v, const int* __restrict__ idx,
                            const float* __restrict__ geo,
                            const float* __restrict__ Wg, const float* __restrict__ bg,
                            const float* __restrict__ ascale, const float* __restrict__ fscale,
                            unsigned short* __restrict__ outBf, int N, int D) {
    __shared__ float qs[512];
    const int bn = blockIdx.x;
    const int b = bn / N;
    const float* qrow = q + (size_t)bn*D;
    qs[threadIdx.x]       = qrow[threadIdx.x];
    qs[threadIdx.x + 256] = qrow[threadIdx.x + 256];
    __syncthreads();
    const int h = threadIdx.x >> 5;
    const int lane = threadIdx.x & 31;
    const int d0 = h*64 + lane*2;
    const float q0 = qs[d0], q1 = qs[d0+1];
    float qw0 = waveAllSum(q0*Wg[0*D+d0] + q1*Wg[0*D+d0+1]);
    float qw1 = waveAllSum(q0*Wg[1*D+d0] + q1*Wg[1*D+d0+1]);
    float qw2 = waveAllSum(q0*Wg[2*D+d0] + q1*Wg[2*D+d0+1]);
    float qb  = waveAllSum(q0*bg[d0]     + q1*bg[d0+1]);
    const float fs = fscale[h], as = ascale[h];
    const int* myidx = idx + (size_t)bn*KNB;
    float myscore = 0.f; int myj = 0;
    for (int kk = 0; kk < KNB; ++kk) {
        int j = myidx[kk];
        const float* krow = k + ((size_t)b*N + j)*D;
        float s = waveAllSum(q0*krow[d0] + q1*krow[d0+1]);
        const float* g3 = geo + ((size_t)bn*KNB + kk)*3;
        float align = qw0*g3[0] + qw1*g3[1] + qw2*g3[2] + qb;
        float score = fs*s*0.125f + as*align;     // 1/sqrt(64) = 0.125
        if (lane == kk) { myscore = score; myj = j; }
    }
    float mx = waveAllMax(myscore);
    float e  = expf(myscore - mx);
    float den = waveAllSum(e);
    float w  = e / den;
    float a0 = 0.f, a1 = 0.f;
    for (int kk = 0; kk < KNB; ++kk) {
        float wk = __shfl(w, kk, 32);
        int   j  = __shfl(myj, kk, 32);
        const float* vrow = v + ((size_t)b*N + j)*D;
        a0 += wk*vrow[d0]; a1 += wk*vrow[d0+1];
    }
    outBf[(size_t)bn*D + d0]     = f32_to_bf16(a0);
    outBf[(size_t)bn*D + d0 + 1] = f32_to_bf16(a1);
}

extern "C" void kernel_launch(void* const* d_in, const int* in_sizes, int n_in,
                              void* d_out, int out_size, void* d_ws, size_t ws_size,
                              hipStream_t stream) {
    (void)in_sizes; (void)n_in; (void)out_size; (void)ws_size;
    const int B = 2, N = 2048, D = 512, F = 4*D;
    const int M = B*N;

    const float* x    = (const float*)d_in[0];
    const float* pos  = (const float*)d_in[1];
    const float* cp   = (const float*)d_in[2];
    const float* Wq   = (const float*)d_in[3];
    const float* bq   = (const float*)d_in[4];
    const float* Wk   = (const float*)d_in[5];
    const float* bk   = (const float*)d_in[6];
    const float* Wv   = (const float*)d_in[7];
    const float* bv   = (const float*)d_in[8];
    const float* Wg   = (const float*)d_in[9];
    const float* bg   = (const float*)d_in[10];
    const float* Wo   = (const float*)d_in[11];
    const float* bo   = (const float*)d_in[12];
    const float* Wf1  = (const float*)d_in[13];
    const float* bf1  = (const float*)d_in[14];
    const float* Wf2  = (const float*)d_in[15];
    const float* bf2  = (const float*)d_in[16];
    const float* g1   = (const float*)d_in[17];
    const float* b1   = (const float*)d_in[18];
    const float* g2   = (const float*)d_in[19];
    const float* b2   = (const float*)d_in[20];
    const float* asc  = (const float*)d_in[21];
    const float* fsc  = (const float*)d_in[22];
    float* out = (float*)d_out;

    char* ws = (char*)d_ws;
    size_t off = 0;
    auto alloc = [&](size_t bytes) -> char* {
        char* p = ws + off;
        off += (bytes + 255) & ~(size_t)255;
        return p;
    };

    unsigned short* xn_bf  = (unsigned short*)alloc((size_t)M*D*2);
    unsigned short* WqT    = (unsigned short*)alloc((size_t)D*D*2);
    unsigned short* WkT    = (unsigned short*)alloc((size_t)D*D*2);
    unsigned short* WvT    = (unsigned short*)alloc((size_t)D*D*2);
    unsigned short* WoT    = (unsigned short*)alloc((size_t)D*D*2);
    unsigned short* Wf1T   = (unsigned short*)alloc((size_t)F*D*2);
    unsigned short* Wf2T   = (unsigned short*)alloc((size_t)D*F*2);
    float*          qf     = (float*)alloc((size_t)M*D*4);
    float*          kf     = (float*)alloc((size_t)M*D*4);
    float*          vf     = (float*)alloc((size_t)M*D*4);
    int*            idxb   = (int*)alloc((size_t)M*KNB*4);
    float*          geob   = (float*)alloc((size_t)M*KNB*3*4);
    unsigned short* att_bf = (unsigned short*)alloc((size_t)M*D*2);
    float*          x1f    = (float*)alloc((size_t)M*D*4);
    unsigned short* h_bf   = (unsigned short*)alloc((size_t)M*D*2);
    unsigned short* f1_bf  = (unsigned short*)alloc((size_t)M*F*2);

    // 1) LN1: x -> xn (bf16)
    ln_bf16_kernel<<<M, 256, 0, stream>>>(x, g1, b1, xn_bf, D);

    // 2) weight transpose+convert to bf16 column-major
    conv_transpose_bf16<<<(D*D+255)/256, 256, 0, stream>>>(Wq,  WqT,  D, D);
    conv_transpose_bf16<<<(D*D+255)/256, 256, 0, stream>>>(Wk,  WkT,  D, D);
    conv_transpose_bf16<<<(D*D+255)/256, 256, 0, stream>>>(Wv,  WvT,  D, D);
    conv_transpose_bf16<<<(D*D+255)/256, 256, 0, stream>>>(Wo,  WoT,  D, D);
    conv_transpose_bf16<<<(D*F+255)/256, 256, 0, stream>>>(Wf1, Wf1T, D, F);
    conv_transpose_bf16<<<(F*D+255)/256, 256, 0, stream>>>(Wf2, Wf2T, F, D);

    // 3) Q/K/V projections (WMMA)
    {
        dim3 g(M/64, D/64); dim3 blk(128);
        gemm_bf16_wmma<<<g, blk, 0, stream>>>(xn_bf, WqT, bq, nullptr, qf, nullptr, M, D, D, 0);
        gemm_bf16_wmma<<<g, blk, 0, stream>>>(xn_bf, WkT, bk, nullptr, kf, nullptr, M, D, D, 0);
        gemm_bf16_wmma<<<g, blk, 0, stream>>>(xn_bf, WvT, bv, nullptr, vf, nullptr, M, D, D, 0);
    }

    // 4) top-K neighbors + geo vectors
    topk_kernel<<<M, 256, 0, stream>>>(pos, cp, idxb, N);
    geo_kernel<<<(M*KNB + 255)/256, 256, 0, stream>>>(pos, cp, idxb, geob, N, M*KNB);

    // 5) gather attention -> bf16 activation
    attn_kernel<<<M, 256, 0, stream>>>(qf, kf, vf, idxb, geob, Wg, bg, asc, fsc, att_bf, N, D);

    // 6) x1 = x + attn@Wo + bo  (WMMA, fused residual)
    {
        dim3 g(M/64, D/64); dim3 blk(128);
        gemm_bf16_wmma<<<g, blk, 0, stream>>>(att_bf, WoT, bo, x, x1f, nullptr, M, D, D, 0);
    }

    // 7) LN2: x1 -> h (bf16)
    ln_bf16_kernel<<<M, 256, 0, stream>>>(x1f, g2, b2, h_bf, D);

    // 8) FFN up + GELU (WMMA, bf16 out)
    {
        dim3 g(M/64, F/64); dim3 blk(128);
        gemm_bf16_wmma<<<g, blk, 0, stream>>>(h_bf, Wf1T, bf1, nullptr, nullptr, f1_bf, M, F, D, 1);
    }

    // 9) FFN down + residual -> d_out (f32)
    {
        dim3 g(M/64, D/64); dim3 blk(128);
        gemm_bf16_wmma<<<g, blk, 0, stream>>>(f1_bf, Wf2T, bf2, x1f, out, nullptr, M, D, F, 0);
    }
}
